// DocLSTM_79018808312396
// MI455X (gfx1250) — compile-verified, matrix operand order checked by reading, not run on hardware
//
#include <hip/hip_runtime.h>
#include <hip/hip_bf16.h>
#include <math.h>

// ---------------------------------------------------------------------------
// CDNA5 (gfx1250) implementation of the GAT + retrieval-attention reference.
//  * Dense X@W projections  -> v_wmma_f32_16x16x32_bf16, 4 column-tiles/wave,
//    W pre-packed to WMMA B-layout bf16 (loaded as b128), A as b64 f32 pairs.
//  * Edge softmax aggregation -> ordered-uint atomicMax + f32 atomic adds
//  * Pooling / tiny MHA       -> plain VALU kernels
// ---------------------------------------------------------------------------

typedef __attribute__((ext_vector_type(16))) __bf16 v16bf;
typedef __attribute__((ext_vector_type(8)))  float  v8f;

#define N_BODY 30000
#define N_HEAD 64
#define E_BODY 300000
#define E_HEAD 512
#define GAT_H  2

// ---- helpers --------------------------------------------------------------

__device__ __forceinline__ unsigned f2ord(float f) {
    unsigned u = __float_as_uint(f);
    return (u & 0x80000000u) ? ~u : (u | 0x80000000u);
}
__device__ __forceinline__ float ord2f(unsigned u) {
    return (u & 0x80000000u) ? __uint_as_float(u & 0x7fffffffu)
                             : __uint_as_float(~u);
}
__device__ __forceinline__ void atomAddF(float* p, float v) {
    unsafeAtomicAdd(p, v);   // lowers to global_atomic_add_f32 on gfx1250
}
__device__ __forceinline__ v8f wmma_bf16(v16bf a, v16bf b, v8f c) {
    return __builtin_amdgcn_wmma_f32_16x16x32_bf16(false, a, false, b,
                                                   (short)0, c, false, false);
}

// K-offset pattern inside a 32-wide K tile for 16-bit A operands
// (ISA 7.12.2): lane half (kbase 0/8) is folded into the base pointer, so the
// per-register offsets are the compile-time constants below.
__device__ __forceinline__ constexpr int kOff(int j) {
    return (j < 4) ? 2 * j : 16 + 2 * (j - 4);
}

// ---------------------------------------------------------------------------
// W pre-pack: fp32 W[K,M] -> bf16 WMMA B-layout.
// WP logical layout: [KT][M/16][lane(32)][16 bf16]  (32B per lane record).
// grid = (M/16, KT), block = 32.
// ---------------------------------------------------------------------------
template <int K, int M>
__global__ void pack_w_bf16(const float* __restrict__ W, __bf16* __restrict__ WP) {
    constexpr int CT = M / 16;
    const int ct = blockIdx.x, kt = blockIdx.y, lane = threadIdx.x;
    const int lo = lane & 15;
    const int kb = (lane >> 4) ? 8 : 0;
    const int col = ct * 16 + lo;
    v16bf b{};
#pragma unroll
    for (int j = 0; j < 8; ++j) {
        int k0 = kt * 32 + kb + kOff(j);
        // clamped address + select: no exec-divergent guarded loads
        int k0s = (k0 < K) ? k0 : 0;
        int k1s = (k0 + 1 < K) ? k0 + 1 : 0;
        float w0 = W[(size_t)k0s * M + col];
        float w1 = W[(size_t)k1s * M + col];
        b[2 * j]     = (__bf16)((k0 < K) ? w0 : 0.f);
        b[2 * j + 1] = (__bf16)((k0 + 1 < K) ? w1 : 0.f);
    }
    ((v16bf*)WP)[((size_t)kt * CT + ct) * 32 + lane] = b;
}

// ---------------------------------------------------------------------------
// GEMM: Y[N,M] = X[N,K] @ W[K,M] with bf16 WMMA compute.
// grid = (M/64, N/16), block = 32 (one wave -> 16 rows x 64 cols = 4 tiles).
// Main K loop is unguarded; K%32 tail (K=300 only) uses clamp+select.
// ---------------------------------------------------------------------------
template <int K, int M>
__global__ void gemm_wmma_bf16(const float* __restrict__ X,
                               const __bf16* __restrict__ WP,
                               float* __restrict__ Y) {
    constexpr int CT = M / 16;          // column tiles in W
    constexpr int KF = (K / 32) * 32;   // full K tiles
    const int lane = threadIdx.x;
    const int lo = lane & 15, hi = lane >> 4;
    const int kb = hi ? 8 : 0;
    const int row = blockIdx.y * 16 + lo;
    const int ct0 = blockIdx.x * 4;

    const float* pA = X + (size_t)row * K + kb;
    const v16bf* pB = (const v16bf*)WP + (size_t)ct0 * 32 + lane;

    v8f acc0{}, acc1{}, acc2{}, acc3{};

    for (int k0 = 0; k0 < KF; k0 += 32) {
        __builtin_prefetch(pA + 32, 0, 1);       // global_prefetch_b8
        v16bf a;
#pragma unroll
        for (int j = 0; j < 8; ++j) {
            float2 p = *(const float2*)(pA + kOff(j));  // b64, pairs are contiguous
            a[2 * j]     = (__bf16)p.x;
            a[2 * j + 1] = (__bf16)p.y;
        }
        v16bf b0 = pB[0 * 32];
        v16bf b1 = pB[1 * 32];
        v16bf b2 = pB[2 * 32];
        v16bf b3 = pB[3 * 32];
        acc0 = wmma_bf16(a, b0, acc0);
        acc1 = wmma_bf16(a, b1, acc1);
        acc2 = wmma_bf16(a, b2, acc2);
        acc3 = wmma_bf16(a, b3, acc3);
        pA += 32;
        pB += (size_t)CT * 32;
    }

    if constexpr (KF < K) {                       // tail (e.g. K=300 -> 12)
        constexpr int TR = K - KF;
        v16bf a{};
#pragma unroll
        for (int j = 0; j < 8; ++j) {
            int off = kOff(j);
            bool v0 = (kb + off) < TR;
            bool v1 = (kb + off + 1) < TR;
            float x0 = pA[v0 ? off : 0];          // clamped addr, cndmask value
            float x1 = pA[v1 ? off + 1 : 0];
            a[2 * j]     = (__bf16)(v0 ? x0 : 0.f);
            a[2 * j + 1] = (__bf16)(v1 ? x1 : 0.f);
        }
        v16bf b0 = pB[0 * 32];
        v16bf b1 = pB[1 * 32];
        v16bf b2 = pB[2 * 32];
        v16bf b3 = pB[3 * 32];
        acc0 = wmma_bf16(a, b0, acc0);
        acc1 = wmma_bf16(a, b1, acc1);
        acc2 = wmma_bf16(a, b2, acc2);
        acc3 = wmma_bf16(a, b3, acc3);
    }

    // C/D layout: lane = column (lo), VGPR v = row v + 8*hi
#pragma unroll
    for (int v = 0; v < 8; ++v) {
        int r = blockIdx.y * 16 + v + 8 * hi;
        float* yr = Y + (size_t)r * M + ct0 * 16 + lo;
        yr[0]  = acc0[v];
        yr[16] = acc1[v];
        yr[32] = acc2[v];
        yr[48] = acc3[v];
    }
}

// ---- GAT edge pipeline ----------------------------------------------------
// xp layout: [N][H][F] flat (GEMM output columns = H*F).

__global__ void as_ad_kernel(const float* __restrict__ xp,
                             const float* __restrict__ asrc,
                             const float* __restrict__ adst,
                             float* __restrict__ as_, float* __restrict__ ad_,
                             int N, int H, int F) {
    int i = blockIdx.x * blockDim.x + threadIdx.x;
    if (i >= N * H) return;
    int h = i % H;
    const float* xrow = xp + (size_t)i * F;   // i = n*H+h  ->  offset n*H*F + h*F
    float s = 0.f, d = 0.f;
    for (int f = 0; f < F; ++f) {
        float v = xrow[f];
        s += v * asrc[h * F + f];
        d += v * adst[h * F + f];
    }
    as_[i] = s;
    ad_[i] = d;
}

__global__ void gat_init_kernel(unsigned* __restrict__ mord,
                                float* __restrict__ den,
                                float* __restrict__ acc,
                                int NH, long long NHF) {
    long long i = (long long)blockIdx.x * blockDim.x + threadIdx.x;
    if (i < NHF) acc[i] = 0.f;
    if (i < NH) { mord[i] = f2ord(-INFINITY); den[i] = 0.f; }
}

__global__ void edge_alpha_max_kernel(const float* __restrict__ as_,
                                      const float* __restrict__ ad_,
                                      const int* __restrict__ el,
                                      int E, int N, int H,
                                      float* __restrict__ alpha,
                                      unsigned* __restrict__ mord) {
    int i = blockIdx.x * blockDim.x + threadIdx.x;
    int Etot = E + N;
    if (i >= Etot * H) return;
    int e = i / H, h = i - e * H;
    int s, d;
    if (e < E) { s = el[2 * e]; d = el[2 * e + 1]; }
    else       { s = d = e - E; }                       // self loop
    float x = as_[s * H + h] + ad_[d * H + h];
    float a = (x > 0.f) ? x : 0.2f * x;                 // leaky_relu(0.2)
    alpha[i] = a;
    atomicMax(mord + d * H + h, f2ord(a));
}

__global__ void edge_expsum_kernel(float* __restrict__ alpha,
                                   const unsigned* __restrict__ mord,
                                   const int* __restrict__ el,
                                   int E, int N, int H,
                                   float* __restrict__ den) {
    int i = blockIdx.x * blockDim.x + threadIdx.x;
    int Etot = E + N;
    if (i >= Etot * H) return;
    int e = i / H, h = i - e * H;
    int d;
    if (e < E) d = el[2 * e + 1]; else d = e - E;
    float m  = ord2f(mord[d * H + h]);
    float ex = __expf(alpha[i] - m);
    alpha[i] = ex;                                       // reuse buffer: e-values
    atomAddF(den + d * H + h, ex);
}

__global__ void edge_scatter_kernel(const float* __restrict__ exv,
                                    const float* __restrict__ den,
                                    const float* __restrict__ xp,
                                    const int* __restrict__ el,
                                    int E, int N, int H, int F,
                                    float* __restrict__ acc) {
    int C = F >> 5;                                      // 32-feature chunks
    long long total = (long long)(E + N) * H * C;
    long long i = (long long)blockIdx.x * blockDim.x + threadIdx.x;
    if (i >= total) return;
    int c = (int)(i % C);
    long long eh = i / C;
    int h = (int)(eh % H);
    int e = (int)(eh / H);
    int s, d;
    if (e < E) { s = el[2 * e]; d = el[2 * e + 1]; }
    else       { s = d = e - E; }
    float coef = exv[eh] / fmaxf(den[d * H + h], 1e-16f);
    const float* xs = xp  + ((size_t)s * H + h) * F + c * 32;
    float*       ao = acc + ((size_t)d * H + h) * F + c * 32;
#pragma unroll
    for (int f = 0; f < 32; ++f) atomAddF(ao + f, coef * xs[f]);
}

__global__ void finalize_kernel(const float* __restrict__ acc,
                                const float* __restrict__ bias,
                                float* __restrict__ out,
                                int N, int H, int F) {
    int i = blockIdx.x * blockDim.x + threadIdx.x;
    if (i >= N * F) return;
    int n = i / F, f = i - n * F;
    float s = 0.f;
    for (int h = 0; h < H; ++h) s += acc[((size_t)n * H + h) * F + f];
    out[i] = tanhf(s / (float)H + bias[f]);              // mean over heads + b
}

// ---- pooling --------------------------------------------------------------
// x: [N,32] -> out96 = [max(32) | min(32) | mean(32)]

__global__ void pool_all_kernel(const float* __restrict__ x, int N,
                                float* __restrict__ out96) {
    __shared__ float smx[8][32], smn[8][32], ssm[8][32];
    int t = threadIdx.x, f = t & 31, g = t >> 5;
    float mx = -INFINITY, mn = INFINITY, sm = 0.f;
    for (int n = g; n < N; n += 8) {
        float v = x[(size_t)n * 32 + f];
        mx = fmaxf(mx, v); mn = fminf(mn, v); sm += v;
    }
    smx[g][f] = mx; smn[g][f] = mn; ssm[g][f] = sm;
    __syncthreads();
    if (g == 0) {
        for (int k = 1; k < 8; ++k) {
            mx = fmaxf(mx, smx[k][f]); mn = fminf(mn, smn[k][f]); sm += ssm[k][f];
        }
        out96[f] = mx; out96[32 + f] = mn; out96[64 + f] = sm / (float)N;
    }
}

// mask: [12,N] bool(u8); x: [N,32]; pooled: [12,96]; grid.x = 12
__global__ void masked_pool_kernel(const unsigned char* __restrict__ mask,
                                   const float* __restrict__ x, int N,
                                   float* __restrict__ pooled) {
    __shared__ float smx[8][32], smn[8][32], ssm[8][32];
    __shared__ int   scnt[8];
    int kq = blockIdx.x;
    const unsigned char* mk = mask + (size_t)kq * N;
    int t = threadIdx.x, f = t & 31, g = t >> 5;
    float mx = -INFINITY, mn = INFINITY, sm = 0.f;
    int cnt = 0;
    for (int n = g; n < N; n += 8) {
        if (mk[n]) {
            float v = x[(size_t)n * 32 + f];
            mx = fmaxf(mx, v); mn = fminf(mn, v); sm += v;
            if (f == 0) ++cnt;
        }
    }
    smx[g][f] = mx; smn[g][f] = mn; ssm[g][f] = sm;
    if (f == 0) scnt[g] = cnt;
    __syncthreads();
    if (g == 0) {
        int c = 0;
        for (int k = 0; k < 8; ++k) c += scnt[k];
        for (int k = 1; k < 8; ++k) {
            mx = fmaxf(mx, smx[k][f]); mn = fminf(mn, smn[k][f]); sm += ssm[k][f];
        }
        float* o = pooled + (size_t)kq * 96;
        o[f] = mx; o[32 + f] = mn; o[64 + f] = sm / (float)c;
    }
}

// ---- tiny MHA (B=1, Lq=1, Lk=12, E=96, 4 heads x 24) ---------------------
__global__ void mha_kernel(const float* __restrict__ q96,
                           const float* __restrict__ kv,   // [12,96]
                           const float* __restrict__ wq, const float* __restrict__ bq,
                           const float* __restrict__ wk, const float* __restrict__ bk,
                           const float* __restrict__ wv, const float* __restrict__ bv,
                           const float* __restrict__ wo, const float* __restrict__ bo,
                           int indicator, float* __restrict__ out96) {
    __shared__ float qp[96], kp[12 * 96], vp[12 * 96], aw[4 * 12], o[96];
    int t = threadIdx.x;                                  // 128 threads
    if (t < 96) {
        float s = bq[t];
        for (int i = 0; i < 96; ++i) s += q96[i] * wq[i * 96 + t];
        qp[t] = s;
    }
    for (int idx = t; idx < 12 * 96; idx += 128) {
        int j = idx / 96, c = idx % 96;
        float sk = bk[c], sv = bv[c];
        for (int i = 0; i < 96; ++i) {
            float x = kv[j * 96 + i];
            sk += x * wk[i * 96 + c];
            sv += x * wv[i * 96 + c];
        }
        kp[idx] = sk; vp[idx] = sv;
    }
    __syncthreads();
    if (t < 48) {
        int h = t / 12, j = t % 12;
        float s = 0.f;
        for (int d = 0; d < 24; ++d) s += qp[h * 24 + d] * kp[j * 96 + h * 24 + d];
        s *= 0.2041241452319315f;                         // 1/sqrt(24)
        float sig = 1.f / (1.f + __expf(-s));
        aw[h * 12 + j] = indicator ? (1.f - sig) : sig;
    }
    __syncthreads();
    if (t < 4) {
        int h = t;
        float mx = -INFINITY;
        for (int j = 0; j < 12; ++j) mx = fmaxf(mx, aw[h * 12 + j]);
        float sum = 0.f;
        for (int j = 0; j < 12; ++j) {
            float e = __expf(aw[h * 12 + j] - mx);
            aw[h * 12 + j] = e; sum += e;
        }
        for (int j = 0; j < 12; ++j) aw[h * 12 + j] /= sum;
    }
    __syncthreads();
    if (t < 96) {
        int h = t / 24;
        float s = 0.f;
        for (int j = 0; j < 12; ++j) s += aw[h * 12 + j] * vp[j * 96 + t];
        o[t] = s;
    }
    __syncthreads();
    if (t < 96) {
        float s = bo[t];
        for (int i = 0; i < 96; ++i) s += o[i] * wo[i * 96 + t];
        out96[t] = s;
    }
}

// ---------------------------------------------------------------------------
// Host orchestration
// ---------------------------------------------------------------------------
namespace {
struct GatBufs {
    float *xp, *acc, *as, *ad, *den, *alpha;
    unsigned* mord;
    __bf16* wp;
};

template <int K, int F>
inline void run_gat_layer(const float* X, int N, const int* el, int E,
                          const float* W, const float* asr, const float* ads,
                          const float* bias, float* out,
                          const GatBufs& g, hipStream_t s) {
    constexpr int H = GAT_H, M = H * F;
    constexpr int KT = (K + 31) / 32;

    pack_w_bf16<K, M><<<dim3(M / 16, KT), 32, 0, s>>>(W, g.wp);
    gemm_wmma_bf16<K, M><<<dim3(M / 64, N / 16), 32, 0, s>>>(X, g.wp, g.xp);

    int nh = N * H;
    as_ad_kernel<<<(nh + 255) / 256, 256, 0, s>>>(g.xp, asr, ads, g.as, g.ad, N, H, F);

    long long nhf = (long long)nh * F;
    gat_init_kernel<<<(int)((nhf + 255) / 256), 256, 0, s>>>(g.mord, g.den, g.acc, nh, nhf);

    int etoth = (E + N) * H;
    edge_alpha_max_kernel<<<(etoth + 255) / 256, 256, 0, s>>>(g.as, g.ad, el, E, N, H,
                                                              g.alpha, g.mord);
    edge_expsum_kernel<<<(etoth + 255) / 256, 256, 0, s>>>(g.alpha, g.mord, el, E, N, H,
                                                           g.den);
    long long tot = (long long)(E + N) * H * (F >> 5);
    edge_scatter_kernel<<<(int)((tot + 255) / 256), 256, 0, s>>>(g.alpha, g.den, g.xp,
                                                                 el, E, N, H, F, g.acc);
    finalize_kernel<<<(N * F + 255) / 256, 256, 0, s>>>(g.acc, bias, out, N, H, F);
}
} // namespace

extern "C" void kernel_launch(void* const* d_in, const int* in_sizes, int n_in,
                              void* d_out, int out_size, void* d_ws, size_t ws_size,
                              hipStream_t stream) {
    (void)in_sizes; (void)n_in; (void)out_size; (void)ws_size;

    // --- inputs (setup_inputs dict order, params flattened in insertion order)
    const float* b_feat = (const float*)d_in[0];           // [30000,300]
    const float* h_feat = (const float*)d_in[1];           // [64,300]
    const int*   b_el   = (const int*)d_in[2];             // [300000,2]
    const int*   h_el   = (const int*)d_in[3];             // [512,2]
    const unsigned char* pos_m = (const unsigned char*)d_in[4];  // [12,30000] bool
    const unsigned char* neg_m = (const unsigned char*)d_in[5];
    const float* c1W = (const float*)d_in[6],  *c1as = (const float*)d_in[7],
               * c1ad = (const float*)d_in[8], *c1b  = (const float*)d_in[9];
    const float* c2W = (const float*)d_in[10], *c2as = (const float*)d_in[11],
               * c2ad = (const float*)d_in[12], *c2b = (const float*)d_in[13];
    const float* c3W = (const float*)d_in[14], *c3as = (const float*)d_in[15],
               * c3ad = (const float*)d_in[16], *c3b = (const float*)d_in[17];
    const float* h1W = (const float*)d_in[18], *h1as = (const float*)d_in[19],
               * h1ad = (const float*)d_in[20], *h1b = (const float*)d_in[21];
    const float* h2W = (const float*)d_in[22], *h2as = (const float*)d_in[23],
               * h2ad = (const float*)d_in[24], *h2b = (const float*)d_in[25];
    const float* wq = (const float*)d_in[26], *bq = (const float*)d_in[27];
    const float* wk = (const float*)d_in[28], *bk = (const float*)d_in[29];
    const float* wv = (const float*)d_in[30], *bv = (const float*)d_in[31];
    const float* wo = (const float*)d_in[32], *bo = (const float*)d_in[33];

    float* out = (float*)d_out;   // [head_pooled|pos_rep|neg_rep|global_rep] = 384 f32

    // --- workspace carve (floats) ------------------------------------------
    float* ws = (float*)d_ws;
    size_t o = 0;
    float* XP   = ws + o; o += (size_t)N_BODY * 256;       // max H*F = 256
    float* ACC  = ws + o; o += (size_t)N_BODY * 256;
    float* B1   = ws + o; o += (size_t)N_BODY * 128;
    float* B2   = ws + o; o += (size_t)N_BODY * 64;
    float* B3   = ws + o; o += (size_t)N_BODY * 32;
    float* H1   = ws + o; o += (size_t)N_HEAD * 128;
    float* H2   = ws + o; o += (size_t)N_HEAD * 32;
    float* AS   = ws + o; o += (size_t)N_BODY * GAT_H;
    float* AD   = ws + o; o += (size_t)N_BODY * GAT_H;
    unsigned* MORD = (unsigned*)(ws + o); o += (size_t)N_BODY * GAT_H;
    float* DEN  = ws + o; o += (size_t)N_BODY * GAT_H;
    float* ALPH = ws + o; o += (size_t)(E_BODY + N_BODY) * GAT_H;
    float* POSP = ws + o; o += 12 * 96;
    float* NEGP = ws + o; o += 12 * 96;
    __bf16* WP  = (__bf16*)(ws + o); o += 48 * 1024;       // packed W (<=160KB)

    GatBufs g{XP, ACC, AS, AD, DEN, ALPH, MORD, WP};

    // --- body GAT stack: 300 -> 128 -> 64 -> 32 ----------------------------
    run_gat_layer<300, 128>(b_feat, N_BODY, b_el, E_BODY, c1W, c1as, c1ad, c1b, B1, g, stream);
    run_gat_layer<128,  64>(B1,     N_BODY, b_el, E_BODY, c2W, c2as, c2ad, c2b, B2, g, stream);
    run_gat_layer< 64,  32>(B2,     N_BODY, b_el, E_BODY, c3W, c3as, c3ad, c3b, B3, g, stream);

    // --- head GAT stack: 300 -> 128 -> 32 (reuses the same scratch) --------
    run_gat_layer<300, 128>(h_feat, N_HEAD, h_el, E_HEAD, h1W, h1as, h1ad, h1b, H1, g, stream);
    run_gat_layer<128,  32>(H1,     N_HEAD, h_el, E_HEAD, h2W, h2as, h2ad, h2b, H2, g, stream);

    // --- pooling -----------------------------------------------------------
    pool_all_kernel<<<1, 256, 0, stream>>>(H2, N_HEAD, out);           // head_pooled
    pool_all_kernel<<<1, 256, 0, stream>>>(B3, N_BODY, out + 288);     // global_rep
    masked_pool_kernel<<<12, 256, 0, stream>>>(pos_m, B3, N_BODY, POSP);
    masked_pool_kernel<<<12, 256, 0, stream>>>(neg_m, B3, N_BODY, NEGP);

    // --- retrieval attention ----------------------------------------------
    mha_kernel<<<1, 128, 0, stream>>>(out, POSP, wq, bq, wk, bk, wv, bv, wo, bo,
                                      0, out + 96);                    // pos_rep
    mha_kernel<<<1, 128, 0, stream>>>(out, NEGP, wq, bq, wk, bk, wv, bv, wo, bo,
                                      1, out + 192);                   // neg_rep
}
